// NodeGLAM_26207890440557
// MI455X (gfx1250) — matrix operand order repaired and checked by reading, not compiled, not run on hardware
//
#include <hip/hip_runtime.h>

typedef __attribute__((ext_vector_type(16))) _Float16 v16h;
typedef __attribute__((ext_vector_type(8)))  _Float16 v8h;
typedef __attribute__((ext_vector_type(8)))  float    v8f;

#define NN    50000
#define MPAD  50016          /* 32 * 1563 : padded row count for 32x32 tiles */
#define EE    800000
#define DIN   64
#define HF    128
#define DOUT  16

#define FLAG_RELU 1
#define FLAG_BIAS 2
#define FLAG_ACC  4
#define FLAG_ST16 8

// ---------------------------------------------------------------------------
// Fragment loaders (f16 operands, documented CDNA5 WMMA layouts)
//   A 16x32: lane(hlf,l) -> row l ; elem j   = K(hlf*8 + j)
//                                   elem 8+j = K(hlf*8 + 16 + j)
//   B 32x16 (from transposed weights Bt[n][k]): same K order as A
// ---------------------------------------------------------------------------
static __device__ inline v16h load_a_frag(const _Float16* p) {
  v8h lo = *(const v8h*)p;          // K +0..7
  v8h hi = *(const v8h*)(p + 16);   // K +16..23
  return __builtin_shufflevector(lo, hi, 0,1,2,3,4,5,6,7,8,9,10,11,12,13,14,15);
}
static __device__ inline v16h load_b_frag(const _Float16* p) {
  v8h lo = *(const v8h*)p;          // K +0..7
  v8h hi = *(const v8h*)(p + 16);   // K +16..23
  return __builtin_shufflevector(lo, hi, 0,1,2,3,4,5,6,7,8,9,10,11,12,13,14,15);
}

// ---------------------------------------------------------------------------
// WMMA GEMM, 32x32 tile per wave (2x2 v_wmma_f32_16x16x32_f16 per K-step).
// A is f16 row-major [Mpad, lda]; Bt is f16 *transposed* weights [Nc, ldbt]
// (Bt[n*ldbt+k] = W[k][n]).  Accumulator f32.  Mpad%32==0, Nc%32==0, K%32==0.
// Output either f32 (Cacc) or f16 (D16).  Branch-free inner loop; tile id is
// wave-uniform so EXEC stays all-ones (WMMA requirement).
// ---------------------------------------------------------------------------
__global__ __launch_bounds__(128) void wmma_gemm2x2(
    const _Float16* __restrict__ A, int lda,
    const _Float16* __restrict__ Bt, int ldbt,
    const float* __restrict__ bias,
    float* __restrict__ Cacc, int ldc,
    _Float16* __restrict__ D16, int ldd,
    int Mpad, int Nc, int Kdim, int flags)
{
  const int wavesPerBlock = blockDim.x >> 5;
  int wave = blockIdx.x * wavesPerBlock + (threadIdx.x >> 5);
  int npairs = Nc >> 5;
  int mpairs = Mpad >> 5;
  if (wave >= mpairs * npairs) return;           // wave-uniform
  int mp = wave / npairs;
  int np = wave - mp * npairs;

  int lane = threadIdx.x & 31;
  int hlf  = lane >> 4;
  int l15  = lane & 15;
  int m0   = mp << 5;
  int n0   = np << 5;
  int nA   = n0 + l15;        // column of c*0
  int nB   = nA + 16;         // column of c*1
  int r0   = m0 + hlf * 8;    // row base of c0* ; c1* at +16

  v8f c00, c01, c10, c11;
#pragma unroll
  for (int r = 0; r < 8; ++r) {
    float v00 = 0.f, v01 = 0.f, v10 = 0.f, v11 = 0.f;
    if (flags & FLAG_ACC) {
      const float* cr0 = Cacc + (size_t)(r0 + r) * ldc;
      const float* cr1 = cr0 + (size_t)16 * ldc;
      v00 = cr0[nA]; v01 = cr0[nB]; v10 = cr1[nA]; v11 = cr1[nB];
    }
    if (flags & FLAG_BIAS) {
      float bA = bias[nA], bB = bias[nB];
      v00 += bA; v01 += bB; v10 += bA; v11 += bB;
    }
    c00[r] = v00; c01[r] = v01; c10[r] = v10; c11[r] = v11;
  }

  const _Float16* a0p = A + (size_t)(m0 + l15) * lda + hlf * 8;
  const _Float16* a1p = a0p + (size_t)16 * lda;
  const _Float16* b0p = Bt + (size_t)nA * ldbt + hlf * 8;
  const _Float16* b1p = Bt + (size_t)nB * ldbt + hlf * 8;

  for (int k0 = 0; k0 < Kdim; k0 += 32) {
    v16h a0 = load_a_frag(a0p + k0);
    v16h a1 = load_a_frag(a1p + k0);
    v16h b0 = load_b_frag(b0p + k0);
    v16h b1 = load_b_frag(b1p + k0);
    c00 = __builtin_amdgcn_wmma_f32_16x16x32_f16(false, a0, false, b0, (short)0, c00, false, false);
    c01 = __builtin_amdgcn_wmma_f32_16x16x32_f16(false, a0, false, b1, (short)0, c01, false, false);
    c10 = __builtin_amdgcn_wmma_f32_16x16x32_f16(false, a1, false, b0, (short)0, c10, false, false);
    c11 = __builtin_amdgcn_wmma_f32_16x16x32_f16(false, a1, false, b1, (short)0, c11, false, false);
  }

  if (flags & FLAG_ST16) {
#pragma unroll
    for (int r = 0; r < 8; ++r) {
      float v00 = c00[r], v01 = c01[r], v10 = c10[r], v11 = c11[r];
      if (flags & FLAG_RELU) {
        v00 = fmaxf(v00, 0.f); v01 = fmaxf(v01, 0.f);
        v10 = fmaxf(v10, 0.f); v11 = fmaxf(v11, 0.f);
      }
      _Float16* d0 = D16 + (size_t)(r0 + r) * ldd;
      _Float16* d1 = d0 + (size_t)16 * ldd;
      d0[nA] = (_Float16)v00; d0[nB] = (_Float16)v01;
      d1[nA] = (_Float16)v10; d1[nB] = (_Float16)v11;
    }
  } else {
#pragma unroll
    for (int r = 0; r < 8; ++r) {
      float v00 = c00[r], v01 = c01[r], v10 = c10[r], v11 = c11[r];
      if (flags & FLAG_RELU) {
        v00 = fmaxf(v00, 0.f); v01 = fmaxf(v01, 0.f);
        v10 = fmaxf(v10, 0.f); v11 = fmaxf(v11, 0.f);
      }
      float* d0 = Cacc + (size_t)(r0 + r) * ldc;
      float* d1 = d0 + (size_t)16 * ldc;
      d0[nA] = v00; d0[nB] = v01; d1[nA] = v10; d1[nB] = v11;
    }
  }
}

// ---------------------------------------------------------------------------
// Weight transpose + f32->f16 convert:  Wt[n*K + k] = (n<N) ? W[k*N+n] : 0
// (zero-pad columns so padded GEMM tiles read zeros, e.g. W4 padded 16->32)
// ---------------------------------------------------------------------------
__global__ void wconv(const float* __restrict__ W, _Float16* __restrict__ Wt,
                      int K, int N, int Npad)
{
  int idx = blockIdx.x * blockDim.x + threadIdx.x;
  if (idx >= Npad * K) return;
  int n = idx / K;
  int k = idx - n * K;
  Wt[idx] = (n < N) ? (_Float16)W[(size_t)k * N + n] : (_Float16)0.f;
}

__global__ void b4pad_kernel(const float* __restrict__ b4, float* __restrict__ b4p)
{
  int i = threadIdx.x;   // 32 threads
  b4p[i] = (i < DOUT) ? b4[i] : 0.f;
}

// ---------------------------------------------------------------------------
// Graph preprocessing
// ---------------------------------------------------------------------------
__global__ void deg_kernel(const int* __restrict__ col, float* __restrict__ deg)
{
  int e = blockIdx.x * blockDim.x + threadIdx.x;
  if (e < EE) atomicAdd(&deg[col[e]], 1.0f);
}

__global__ void dis_kernel(const float* __restrict__ deg, float* __restrict__ dis)
{
  int i = blockIdx.x * blockDim.x + threadIdx.x;
  if (i < NN) {
    float d = deg[i];
    dis[i] = (d > 0.0f) ? rsqrtf(fmaxf(d, 1.0f)) : 0.0f;
  }
}

__global__ void norm_kernel(const int* __restrict__ row, const int* __restrict__ col,
                            const float* __restrict__ dis, float* __restrict__ nrm)
{
  int e = blockIdx.x * blockDim.x + threadIdx.x;
  if (e < EE) nrm[e] = dis[row[e]] * dis[col[e]];
}

// packed-f16 atomic add via CDNA5 GLOBAL_ATOMIC_PK_ADD_F16 (no-return form,
// tracked by STOREcnt; s_endpgm implicitly drains it)
static __device__ inline void atomic_pk_add_f16(_Float16* addr, float lo, float hi)
{
  _Float16 l = (_Float16)lo, h = (_Float16)hi;
  unsigned short ul, uh;
  __builtin_memcpy(&ul, &l, 2);
  __builtin_memcpy(&uh, &h, 2);
  unsigned packed = (unsigned)ul | ((unsigned)uh << 16);
  asm volatile("global_atomic_pk_add_f16 %0, %1, off"
               :: "v"(addr), "v"(packed) : "memory");
}

// dst[col[e], :] += norm[e] * src[row[e], :]  over H=128 f16 (64 half2/edge)
__global__ void prop_kernel(const _Float16* __restrict__ src, _Float16* __restrict__ dst,
                            const int* __restrict__ row, const int* __restrict__ col,
                            const float* __restrict__ nrm)
{
  long long idx = (long long)blockIdx.x * blockDim.x + threadIdx.x;
  if (idx >= (long long)EE * (HF / 2)) return;
  int e = (int)(idx >> 6);
  int f = (int)(idx & 63);
  float w = nrm[e];
  union { unsigned u; _Float16 h[2]; } sv;
  sv.u = ((const unsigned*)src)[(size_t)row[e] * (HF / 2) + f];
  atomic_pk_add_f16(dst + (size_t)col[e] * HF + 2 * f,
                    w * (float)sv.h[0], w * (float)sv.h[1]);
}

// ---------------------------------------------------------------------------
// BatchNorm (training-mode batch statistics)
// ---------------------------------------------------------------------------
__global__ void bn_stats(const float* __restrict__ x,
                         float* __restrict__ gsum, float* __restrict__ gsq)
{
  __shared__ float ss[256];
  __shared__ float sq[256];
  int t = threadIdx.x;
  int c = t & 63;
  int r0 = blockIdx.x * 1024 + (t >> 6);
  int rend = min(blockIdx.x * 1024 + 1024, NN);
  float s = 0.0f, s2 = 0.0f;
  for (int r = r0; r < rend; r += 4) {
    float v = x[(size_t)r * DIN + c];
    s += v; s2 += v * v;
  }
  ss[t] = s; sq[t] = s2;
  __syncthreads();
  if (t < 64) {
    float a = ss[t] + ss[t + 64] + ss[t + 128] + ss[t + 192];
    float b = sq[t] + sq[t + 64] + sq[t + 128] + sq[t + 192];
    atomicAdd(&gsum[t], a);
    atomicAdd(&gsq[t], b);
  }
}

__global__ void bn_finalize(const float* __restrict__ gsum, const float* __restrict__ gsq,
                            const float* __restrict__ gamma, const float* __restrict__ beta,
                            float* __restrict__ scale, float* __restrict__ shift)
{
  int c = threadIdx.x;   // 64 threads
  float inv_n = 1.0f / (float)NN;
  float mu  = gsum[c] * inv_n;
  float var = gsq[c] * inv_n - mu * mu;
  float sc  = gamma[c] * rsqrtf(var + 1e-5f);
  scale[c] = sc;
  shift[c] = beta[c] - mu * sc;
}

__global__ void xn_apply(const float* __restrict__ x, const float* __restrict__ scale,
                         const float* __restrict__ shift, _Float16* __restrict__ xn16)
{
  long long i = (long long)blockIdx.x * blockDim.x + threadIdx.x;
  if (i < (long long)NN * DIN) {
    int c = (int)(i & 63);
    xn16[i] = (_Float16)(x[i] * scale[c] + shift[c]);
  }
}

// ---------------------------------------------------------------------------
// Softmax over 16 logits; logits live in lbuf [MPAD,32], output [NN,16]
// ---------------------------------------------------------------------------
__global__ void softmax16(const float* __restrict__ lbuf, float* __restrict__ out)
{
  int i = blockIdx.x * blockDim.x + threadIdx.x;
  if (i >= NN) return;
  const float* p = lbuf + (size_t)i * 32;
  float* q = out + (size_t)i * DOUT;
  float v[DOUT];
  float m = -3.402823e38f;
#pragma unroll
  for (int j = 0; j < DOUT; ++j) { v[j] = p[j]; m = fmaxf(m, v[j]); }
  float s = 0.0f;
#pragma unroll
  for (int j = 0; j < DOUT; ++j) { v[j] = __expf(v[j] - m); s += v[j]; }
  float inv = 1.0f / s;
#pragma unroll
  for (int j = 0; j < DOUT; ++j) q[j] = v[j] * inv;
}

// ---------------------------------------------------------------------------
// Host side
// ---------------------------------------------------------------------------
static inline void launch_gemm(const _Float16* A, int lda, const _Float16* Bt, int ldbt,
                               const float* bias, float* Cacc, int ldc,
                               _Float16* D16, int ldd,
                               int Nc, int K, int flags, hipStream_t s)
{
  int waves  = (MPAD >> 5) * (Nc >> 5);
  int blocks = (waves + 3) / 4;                 // 4 waves per block
  wmma_gemm2x2<<<blocks, 128, 0, s>>>(A, lda, Bt, ldbt, bias, Cacc, ldc,
                                      D16, ldd, MPAD, Nc, K, flags);
}

extern "C" void kernel_launch(void* const* d_in, const int* in_sizes, int n_in,
                              void* d_out, int out_size, void* d_ws, size_t ws_size,
                              hipStream_t stream)
{
  (void)in_sizes; (void)n_in; (void)out_size; (void)ws_size;

  const float* x        = (const float*)d_in[0];
  const int*   row      = (const int*)d_in[1];
  const int*   col      = row + EE;
  const float* bn_gamma = (const float*)d_in[2];
  const float* bn_beta  = (const float*)d_in[3];
  const float* W1       = (const float*)d_in[4];
  const float* b1       = (const float*)d_in[5];
  const float* tagW1    = (const float*)d_in[6];
  const float* tagb1    = (const float*)d_in[7];
  const float* W2       = (const float*)d_in[8];
  const float* b2       = (const float*)d_in[9];
  const float* tagW2    = (const float*)d_in[10];
  const float* tagb2    = (const float*)d_in[11];
  const float* W3       = (const float*)d_in[12];
  const float* b3       = (const float*)d_in[13];
  const float* W4       = (const float*)d_in[14];
  const float* b4       = (const float*)d_in[15];
  float* out = (float*)d_out;

  // ---- workspace carve-up (16B-aligned blocks) ----------------------------
  char* base = (char*)d_ws;
  auto takeF = [&](size_t n) { float* p = (float*)base;
                               base += ((n * 4 + 63) & ~(size_t)63); return p; };
  auto takeH = [&](size_t n) { _Float16* p = (_Float16*)base;
                               base += ((n * 2 + 63) & ~(size_t)63); return p; };
  float* deg  = takeF(NN);
  float* dis  = takeF(NN);
  float* nrm  = takeF(EE);
  float* gsum = takeF(128);            // gsum[0:64] + gsq[64:128], one memset
  float* gsq  = gsum + 64;
  float* scl  = takeF(64);
  float* shf  = takeF(64);
  float* b4p  = takeF(32);
  float* acc  = takeF((size_t)MPAD * HF);
  float* lbuf = takeF((size_t)MPAD * 32);
  _Float16* xn16 = takeH((size_t)MPAD * DIN);
  _Float16* hA16 = takeH((size_t)MPAD * HF);
  _Float16* hB16 = takeH((size_t)MPAD * HF);
  _Float16* hC16 = takeH((size_t)MPAD * HF);
  _Float16* W1t  = takeH(128 * 64);
  _Float16* W2t  = takeH(128 * 128);
  _Float16* W3tT = takeH(128 * 64);
  _Float16* W3tB = takeH(128 * 128);
  _Float16* W4t  = takeH(32 * 128);
  _Float16* tW1t = takeH(4 * 128 * 128);
  _Float16* tW2t = takeH(4 * 128 * 128);

  const size_t hBytes = (size_t)MPAD * HF * sizeof(_Float16);
  const long long propThreads = (long long)EE * (HF / 2);
  const int propBlocks = (int)((propThreads + 255) / 256);

  // ---- weight conversion (transpose, f16, pad) ----------------------------
  wconv<<<(128 * 64 + 255) / 256, 256, 0, stream>>>(W1, W1t, 64, 128, 128);
  wconv<<<(128 * 128 + 255) / 256, 256, 0, stream>>>(W2, W2t, 128, 128, 128);
  wconv<<<(128 * 64 + 255) / 256, 256, 0, stream>>>(W3, W3tT, 64, 128, 128);
  wconv<<<(128 * 128 + 255) / 256, 256, 0, stream>>>(W3 + (size_t)DIN * HF, W3tB, 128, 128, 128);
  wconv<<<(32 * 128 + 255) / 256, 256, 0, stream>>>(W4, W4t, 128, 16, 32);
  for (int k = 0; k < 4; ++k) {
    wconv<<<(128 * 128 + 255) / 256, 256, 0, stream>>>(tagW1 + (size_t)k * HF * HF,
                                                       tW1t + (size_t)k * HF * HF, 128, 128, 128);
    wconv<<<(128 * 128 + 255) / 256, 256, 0, stream>>>(tagW2 + (size_t)k * HF * HF,
                                                       tW2t + (size_t)k * HF * HF, 128, 128, 128);
  }
  b4pad_kernel<<<1, 32, 0, stream>>>(b4, b4p);

  // ---- gcn norm -----------------------------------------------------------
  hipMemsetAsync(deg, 0, NN * sizeof(float), stream);
  hipMemsetAsync(gsum, 0, 128 * sizeof(float), stream);
  deg_kernel<<<(EE + 255) / 256, 256, 0, stream>>>(col, deg);
  dis_kernel<<<(NN + 255) / 256, 256, 0, stream>>>(deg, dis);
  norm_kernel<<<(EE + 255) / 256, 256, 0, stream>>>(row, col, dis, nrm);

  // ---- batchnorm -> xn16 --------------------------------------------------
  bn_stats<<<(NN + 1023) / 1024, 256, 0, stream>>>(x, gsum, gsq);
  bn_finalize<<<1, 64, 0, stream>>>(gsum, gsq, bn_gamma, bn_beta, scl, shf);
  xn_apply<<<(int)(((long long)NN * DIN + 255) / 256), 256, 0, stream>>>(x, scl, shf, xn16);

  // ---- h = relu(xn @ W1 + b1) -> hA16 -------------------------------------
  launch_gemm(xn16, DIN, W1t, DIN, b1, nullptr, 0, hA16, HF,
              HF, DIN, FLAG_BIAS | FLAG_RELU | FLAG_ST16, stream);

  // ---- TAGConv 1 (input hA16, f32 accumulator acc, output hC16) -----------
  launch_gemm(hA16, HF, tW1t + 0 * HF * HF, HF, tagb1, acc, HF, nullptr, 0,
              HF, HF, FLAG_BIAS, stream);
  hipMemsetAsync(hB16, 0, hBytes, stream);
  prop_kernel<<<propBlocks, 256, 0, stream>>>(hA16, hB16, row, col, nrm);
  launch_gemm(hB16, HF, tW1t + 1 * HF * HF, HF, nullptr, acc, HF, nullptr, 0,
              HF, HF, FLAG_ACC, stream);
  hipMemsetAsync(hA16, 0, hBytes, stream);
  prop_kernel<<<propBlocks, 256, 0, stream>>>(hB16, hA16, row, col, nrm);
  launch_gemm(hA16, HF, tW1t + 2 * HF * HF, HF, nullptr, acc, HF, nullptr, 0,
              HF, HF, FLAG_ACC, stream);
  hipMemsetAsync(hB16, 0, hBytes, stream);
  prop_kernel<<<propBlocks, 256, 0, stream>>>(hA16, hB16, row, col, nrm);
  launch_gemm(hB16, HF, tW1t + 3 * HF * HF, HF, nullptr, acc, HF, hC16, HF,
              HF, HF, FLAG_ACC | FLAG_RELU | FLAG_ST16, stream);

  // ---- h = relu(h @ W2 + b2) -> hA16 --------------------------------------
  launch_gemm(hC16, HF, W2t, HF, b2, nullptr, 0, hA16, HF,
              HF, HF, FLAG_BIAS | FLAG_RELU | FLAG_ST16, stream);

  // ---- TAGConv 2 (input hA16, output hC16) --------------------------------
  launch_gemm(hA16, HF, tW2t + 0 * HF * HF, HF, tagb2, acc, HF, nullptr, 0,
              HF, HF, FLAG_BIAS, stream);
  hipMemsetAsync(hB16, 0, hBytes, stream);
  prop_kernel<<<propBlocks, 256, 0, stream>>>(hA16, hB16, row, col, nrm);
  launch_gemm(hB16, HF, tW2t + 1 * HF * HF, HF, nullptr, acc, HF, nullptr, 0,
              HF, HF, FLAG_ACC, stream);
  hipMemsetAsync(hA16, 0, hBytes, stream);
  prop_kernel<<<propBlocks, 256, 0, stream>>>(hB16, hA16, row, col, nrm);
  launch_gemm(hA16, HF, tW2t + 2 * HF * HF, HF, nullptr, acc, HF, nullptr, 0,
              HF, HF, FLAG_ACC, stream);
  hipMemsetAsync(hB16, 0, hBytes, stream);
  prop_kernel<<<propBlocks, 256, 0, stream>>>(hA16, hB16, row, col, nrm);
  launch_gemm(hB16, HF, tW2t + 3 * HF * HF, HF, nullptr, acc, HF, hC16, HF,
              HF, HF, FLAG_ACC | FLAG_RELU | FLAG_ST16, stream);

  // ---- a = relu([xn, h] @ W3 + b3) -> hA16 (two accumulated GEMMs) --------
  launch_gemm(xn16, DIN, W3tT, DIN, b3, acc, HF, nullptr, 0,
              HF, DIN, FLAG_BIAS, stream);
  launch_gemm(hC16, HF, W3tB, HF, nullptr, acc, HF, hA16, HF,
              HF, HF, FLAG_ACC | FLAG_RELU | FLAG_ST16, stream);

  // ---- logits (Nc padded 16->32) -> lbuf, then softmax -> d_out -----------
  launch_gemm(hA16, HF, W4t, HF, b4p, lbuf, 32, nullptr, 0,
              32, HF, FLAG_BIAS, stream);
  softmax16<<<(NN + 255) / 256, 256, 0, stream>>>(lbuf, out);
}
